// CrossAttention_1692217114779
// MI455X (gfx1250) — matrix operand order repaired and checked by reading, not compiled, not run on hardware
//
#include <hip/hip_runtime.h>

// ---------------------------------------------------------------------------
// CrossAttention for MI455X (gfx1250, wave32, WMMA)
//   q = x @ Wq ; k = ctx @ Wk ; v = ctx @ Wv        (f16 WMMA GEMMs, f32 acc)
//   per-token 8x8 head attention + softmax          (f32 VALU, wave shuffles)
//   out = attn @ Wo + bo                            (f16 WMMA GEMM, f32 out)
// GEMM: weight slab (128 cols x 512 K, 128KB) staged in LDS via async-to-LDS
// copies (ASYNCcnt), WMMA fed by ds_load_b128; A streamed from global/L2.
// ---------------------------------------------------------------------------

typedef __attribute__((ext_vector_type(16))) _Float16 v16h;
typedef __attribute__((ext_vector_type(8)))  _Float16 v8h;
typedef __attribute__((ext_vector_type(8)))  float    v8f;
typedef __attribute__((ext_vector_type(4)))  int      v4i;

#define KDIM 512
#define NDIM 512

#if defined(__has_builtin)
#if __has_builtin(__builtin_amdgcn_global_load_async_to_lds_b128)
#define HAVE_ASYNC_LDS 1
#endif
#endif

static __device__ __forceinline__ void wait_async_all() {
#if defined(__has_builtin) && __has_builtin(__builtin_amdgcn_s_wait_asynccnt)
  __builtin_amdgcn_s_wait_asynccnt(0);
#elif defined(HAVE_ASYNC_LDS)
  asm volatile("s_wait_asynccnt 0" ::: "memory");
#endif
}

static __device__ __forceinline__ v16h cat8h(v8h lo, v8h hi) {
  return __builtin_shufflevector(lo, hi, 0, 1, 2, 3, 4, 5, 6, 7,
                                 8, 9, 10, 11, 12, 13, 14, 15);
}

// -------------------- fp32 -> f16 elementwise convert ----------------------
__global__ __launch_bounds__(256) void cvt_f16(const float* __restrict__ in,
                                               _Float16* __restrict__ out,
                                               size_t n) {
  size_t i = ((size_t)blockIdx.x * blockDim.x + threadIdx.x) * 8;
  if (i + 8 <= n) {
    float4 a = *(const float4*)(in + i);
    float4 b = *(const float4*)(in + i + 4);
    v8h o;
    o[0] = (_Float16)a.x; o[1] = (_Float16)a.y;
    o[2] = (_Float16)a.z; o[3] = (_Float16)a.w;
    o[4] = (_Float16)b.x; o[5] = (_Float16)b.y;
    o[6] = (_Float16)b.z; o[7] = (_Float16)b.w;
    *(v8h*)(out + i) = o;
  }
}

// ---------------- fp32 (KxN) weight -> f16 transposed (NxK) ----------------
__global__ __launch_bounds__(256) void cvt_wt(const float* __restrict__ W,
                                              _Float16* __restrict__ Wt) {
  int idx = blockIdx.x * blockDim.x + threadIdx.x; // index into Wt = n*512 + k
  int nn = idx >> 9;
  int k  = idx & 511;
  Wt[idx] = (_Float16)W[k * NDIM + nn];
}

// ---------------------- WMMA GEMM: C = A(MxK) * B(KxN) ---------------------
// A  : f16 row-major MxK ; Bt : f16 row-major NxK (B transposed)
// Workgroup: 128x128 tile, 8 waves, wave tile 32x64 (2x4 WMMA tiles).
// Bt slab for the block (128 rows x 512 halves) lives in LDS.
template <bool F32OUT>
__global__ __launch_bounds__(256) void gemm512_wmma(
    const _Float16* __restrict__ A,
    const _Float16* __restrict__ Bt,
    _Float16* __restrict__ Ch,
    float* __restrict__ Cf,
    const float* __restrict__ bias) {
  constexpr int BPAD = 520;               // 512 + 8 halves: bank-conflict-free
  __shared__ _Float16 Bs[128 * BPAD];     // 130 KB of the 320 KB WGP LDS

  const int tid  = threadIdx.x;
  const int lane = tid & 31;
  const int wave = tid >> 5;
  const int l15  = lane & 15;
  const int lhi  = lane >> 4;                       // 0 or 1
  const int rowBase  = blockIdx.y * 128 + (wave >> 1) * 32;
  const int colBlock = blockIdx.x * 128;
  const int wcol     = (wave & 1) * 64;             // wave's column offset in block

  // ---- stage the block's 128 weight rows into LDS (async DMA path) ----
  {
    const _Float16* gsrc = Bt + (size_t)colBlock * KDIM;
    for (int c = tid; c < 8192; c += 256) {         // 8192 x 16B chunks
      int row = c >> 6;                             // 64 chunks per 512-half row
      int off = (c & 63) * 8;
      const _Float16* g = gsrc + (size_t)row * KDIM + off;
      _Float16* l = &Bs[row * BPAD + off];
#ifdef HAVE_ASYNC_LDS
      __builtin_amdgcn_global_load_async_to_lds_b128(
          (__attribute__((address_space(1))) v4i*)g,
          (__attribute__((address_space(3))) v4i*)l, 0, 0);
#else
      *(v8h*)l = *(const v8h*)g;
#endif
    }
  }
  wait_async_all();
  __syncthreads();

  v8f acc[2][4] = {};   // zero-init accumulators

  // A fragment base: lane (0-15) -> row M=l15, lane-hi -> K sub-block +8
  const _Float16* aptr0 = A + (size_t)(rowBase + l15) * KDIM + lhi * 8;
  const _Float16* aptr1 = aptr0 + (size_t)16 * KDIM;

  for (int kk = 0; kk < KDIM; kk += 32) {
    if (kk + 32 < KDIM) {   // pull next A K-slab toward the caches
      __builtin_prefetch(aptr0 + kk + 32, 0, 1);
      __builtin_prefetch(aptr1 + kk + 32, 0, 1);
    }
    // A fragments (16-bit A layout: V0-3 = K[base..base+7], V4-7 = +16)
    v16h af[2];
#pragma unroll
    for (int m = 0; m < 2; ++m) {
      const _Float16* ap = (m == 0 ? aptr0 : aptr1) + kk;
      af[m] = cat8h(*(const v8h*)(ap), *(const v8h*)(ap + 16));
    }
    // B fragments from LDS: K[base..base+7] and K[base+16..base+23] of one output column
    v16h bf[4];
#pragma unroll
    for (int n = 0; n < 4; ++n) {
      const _Float16* bp = &Bs[(wcol + n * 16 + l15) * BPAD + lhi * 8 + kk];
      bf[n] = cat8h(*(const v8h*)(bp), *(const v8h*)(bp + 16));
    }
#pragma unroll
    for (int m = 0; m < 2; ++m)
#pragma unroll
      for (int n = 0; n < 4; ++n)
        acc[m][n] = __builtin_amdgcn_wmma_f32_16x16x32_f16(
            false, af[m], false, bf[n], (short)0, acc[m][n], false, false);
  }

  // C/D layout: VGPR r, lanes 0-15 -> M=r, N=lane; lanes 16-31 -> M=8+r
#pragma unroll
  for (int m = 0; m < 2; ++m) {
    int r0 = rowBase + m * 16 + lhi * 8;
#pragma unroll
    for (int n = 0; n < 4; ++n) {
      int c = colBlock + wcol + n * 16 + l15;
#pragma unroll
      for (int r = 0; r < 8; ++r) {
        size_t idx = (size_t)(r0 + r) * NDIM + c;
        if (F32OUT) Cf[idx] = acc[m][n][r] + bias[c];
        else        Ch[idx] = (_Float16)acc[m][n][r];
      }
    }
  }
}

// ------------- per-token 8x8 head attention (one wave per token) -----------
// scores[h][g] = q[h,:] . k[g,:]  (64-dim), softmax over g, out = P @ v
__global__ __launch_bounds__(256) void attn_heads(
    const _Float16* __restrict__ qh,
    const _Float16* __restrict__ kh,
    const _Float16* __restrict__ vh,
    _Float16* __restrict__ oh, int ntok) {
  int wave = threadIdx.x >> 5;
  int lane = threadIdx.x & 31;
  int t = blockIdx.x * 8 + wave;
  if (t >= ntok) return;
  const _Float16* qrow = qh + (size_t)t * 512;
  const _Float16* krow = kh + (size_t)t * 512;
  const _Float16* vrow = vh + (size_t)t * 512;
  int h = lane >> 2;       // head handled by this group of 4 lanes
  int j = lane & 3;        // position within the group
  int g0 = j * 2, g1 = g0 + 1;
  const _Float16* qp = qrow + h * 64;
  const _Float16* k0 = krow + g0 * 64;
  const _Float16* k1 = krow + g1 * 64;
  float s0 = 0.f, s1 = 0.f;
#pragma unroll 8
  for (int d = 0; d < 64; ++d) {
    float qv = (float)qp[d];
    s0 += qv * (float)k0[d];
    s1 += qv * (float)k1[d];
  }
  // softmax over the 8 scores of row h (2 per lane, across 4 lanes)
  float m = fmaxf(s0, s1);
  for (int off = 1; off < 4; off <<= 1) m = fmaxf(m, __shfl_xor(m, off, 4));
  float p0 = __expf(s0 - m), p1 = __expf(s1 - m);
  float sum = p0 + p1;
  for (int off = 1; off < 4; off <<= 1) sum += __shfl_xor(sum, off, 4);
  float inv = 1.f / sum;
  p0 *= inv; p1 *= inv;
  // broadcast all 8 probabilities of this head across the 4-lane group
  float pg[8];
#pragma unroll
  for (int jj = 0; jj < 4; ++jj) {
    pg[2 * jj]     = __shfl(p0, jj, 4);
    pg[2 * jj + 1] = __shfl(p1, jj, 4);
  }
  // each lane produces 16 of the 64 output features of head h
  int dbase = j * 16;
#pragma unroll
  for (int dd = 0; dd < 16; ++dd) {
    int d = dbase + dd;
    float acc = 0.f;
#pragma unroll
    for (int g = 0; g < 8; ++g) acc += pg[g] * (float)vrow[g * 64 + d];
    oh[(size_t)t * 512 + h * 64 + d] = (_Float16)acc;
  }
}

// ---------------------------------------------------------------------------
extern "C" void kernel_launch(void* const* d_in, const int* in_sizes, int n_in,
                              void* d_out, int out_size, void* d_ws, size_t ws_size,
                              hipStream_t stream) {
  (void)in_sizes; (void)n_in; (void)out_size; (void)ws_size;
  const float* x   = (const float*)d_in[0];
  const float* ctx = (const float*)d_in[1];
  const float* Wq  = (const float*)d_in[2];
  const float* Wk  = (const float*)d_in[3];
  const float* Wv  = (const float*)d_in[4];
  const float* Wo  = (const float*)d_in[5];
  const float* bo  = (const float*)d_in[6];
  float* out = (float*)d_out;

  const size_t M    = 32768;                       // 8 * 4096 tokens
  const size_t NACT = M * 512;                     // activation elements
  const size_t ACT  = NACT * sizeof(_Float16);     // 32 MB per activation

  char* ws = (char*)d_ws;
  _Float16* xh  = (_Float16*)(ws + 0 * ACT);
  _Float16* chx = (_Float16*)(ws + 1 * ACT);
  _Float16* qh  = (_Float16*)(ws + 2 * ACT);
  _Float16* kh  = (_Float16*)(ws + 3 * ACT);
  _Float16* vh  = (_Float16*)(ws + 4 * ACT);
  _Float16* Wqt = (_Float16*)(ws + 5 * ACT);
  _Float16* Wkt = Wqt + 512 * 512;
  _Float16* Wvt = Wkt + 512 * 512;
  _Float16* Wot = Wvt + 512 * 512;
  _Float16* ah  = xh;   // reuse: x is consumed before attention runs

  // 1) convert activations + weights (weights transposed to NxK)
  cvt_f16<<<NACT / (8 * 256), 256, 0, stream>>>(x,   xh,  NACT);
  cvt_f16<<<NACT / (8 * 256), 256, 0, stream>>>(ctx, chx, NACT);
  cvt_wt<<<(512 * 512) / 256, 256, 0, stream>>>(Wq, Wqt);
  cvt_wt<<<(512 * 512) / 256, 256, 0, stream>>>(Wk, Wkt);
  cvt_wt<<<(512 * 512) / 256, 256, 0, stream>>>(Wv, Wvt);
  cvt_wt<<<(512 * 512) / 256, 256, 0, stream>>>(Wo, Wot);

  // 2) q/k/v projections via WMMA
  dim3 gg(NDIM / 128, M / 128);
  gemm512_wmma<false><<<gg, 256, 0, stream>>>(xh,  Wqt, qh, nullptr, nullptr);
  gemm512_wmma<false><<<gg, 256, 0, stream>>>(chx, Wkt, kh, nullptr, nullptr);
  gemm512_wmma<false><<<gg, 256, 0, stream>>>(chx, Wvt, vh, nullptr, nullptr);

  // 3) per-token head attention
  attn_heads<<<M / 8, 256, 0, stream>>>(qh, kh, vh, ah, (int)M);

  // 4) output projection + bias (f32 result)
  gemm512_wmma<true><<<gg, 256, 0, stream>>>(ah, Wot, nullptr, out, bo);
}
